// FusionModel_88424786690202
// MI455X (gfx1250) — matrix operand order repaired and checked by reading, not compiled, not run on hardware
//
#include <hip/hip_runtime.h>
#include <hip/hip_bf16.h>

// ---------------------------------------------------------------------------
// Problem constants (match the reference)
// ---------------------------------------------------------------------------
#define N_NODES 100000
#define N_EDGES 1600000
#define IN_CH   128
#define N_GRAPHS 16
#define OUT_CH  64            // SAGE output channels
#define KCAT    256           // [mean_nb(128) | x(128)]

// Workspace layout (bytes, all 16B aligned)
#define OFF_AGG   ((size_t)0)                               // N*128 f32  = 51.2 MB
#define OFF_DEG   ((size_t)51200000)                        // N f32      = 0.4 MB
#define OFF_XBF   ((size_t)51600000)                        // N*128 bf16 = 25.6 MB
#define OFF_WCAT  ((size_t)77200000)                        // 64*256 bf16= 32 KB
#define OFF_GSUM  ((size_t)77232768)                        // 16*64 f32
#define OFF_GCNT  ((size_t)77236864)                        // 16 f32

typedef __attribute__((ext_vector_type(16))) __bf16 v16bf;
typedef __attribute__((ext_vector_type(8)))  __bf16 v8bf;
typedef __attribute__((ext_vector_type(4)))  __bf16 v4bf;
typedef __attribute__((ext_vector_type(8)))  float  v8f;

// ---------------------------------------------------------------------------
// Kernel 0: zero accumulators, convert x -> bf16, build Wcat = [W_l | W_r] bf16
// ---------------------------------------------------------------------------
__global__ void k0_init(const float* __restrict__ x,
                        const float* __restrict__ W_l,
                        const float* __restrict__ W_r,
                        float* __restrict__ agg,
                        float* __restrict__ deg,
                        __bf16* __restrict__ xbf,
                        __bf16* __restrict__ Wcat,
                        float* __restrict__ gsum,
                        float* __restrict__ gcnt) {
    const int i = blockIdx.x * 256 + threadIdx.x;
    if (i < N_NODES * IN_CH) {
        agg[i] = 0.0f;
        xbf[i] = (__bf16)x[i];
    }
    if (i < N_NODES)      deg[i]  = 0.0f;
    if (i < OUT_CH * KCAT) {
        const int o = i >> 8, k = i & 255;
        const float w = (k < IN_CH) ? W_l[o * IN_CH + k] : W_r[o * IN_CH + (k - IN_CH)];
        Wcat[i] = (__bf16)w;
    }
    if (i < N_GRAPHS * OUT_CH) gsum[i] = 0.0f;
    if (i < N_GRAPHS)          gcnt[i] = 0.0f;
}

// ---------------------------------------------------------------------------
// Kernel 1: edge scatter. One wave-lane handles (edge, 4 channels).
// bf16 gather (L2-resident), fp32 atomic accumulation.
// ---------------------------------------------------------------------------
__global__ void k1_edges(const __bf16* __restrict__ xbf,
                         const int* __restrict__ edge_index,
                         float* __restrict__ agg,
                         float* __restrict__ deg) {
    const long long t = (long long)blockIdx.x * 256 + threadIdx.x;
    const int e = (int)(t >> 5);
    if (e >= N_EDGES) return;
    const int lane = (int)(t & 31);
    const int src = edge_index[e];
    const int dst = edge_index[N_EDGES + e];

    const v4bf m = *(const v4bf*)(xbf + (size_t)src * IN_CH + lane * 4);
    float* ad = agg + (size_t)dst * IN_CH + lane * 4;
    atomicAdd(ad + 0, (float)m[0]);
    atomicAdd(ad + 1, (float)m[1]);
    atomicAdd(ad + 2, (float)m[2]);
    atomicAdd(ad + 3, (float)m[3]);
    if (lane == 0) atomicAdd(deg + dst, 1.0f);
}

// Kernel 1b: per-graph node counts
__global__ void k1b_gcnt(const int* __restrict__ batch, float* __restrict__ gcnt) {
    const int i = blockIdx.x * 256 + threadIdx.x;
    if (i < N_NODES) atomicAdd(&gcnt[batch[i]], 1.0f);
}

// ---------------------------------------------------------------------------
// Kernel 2: fused SAGE linear via WMMA bf16 (K=256) + ReLU + mean-pool scatter.
// Block = 128 threads (4 waves). Block handles 16 nodes; wave w handles
// output-channel tile [w*16, w*16+16). A tile staged in LDS (bf16, padded rows).
// ---------------------------------------------------------------------------
#define AROW 264   // bf16 elements per LDS row (528 B: 16B-aligned, bank-spread)

__global__ void k2_wmma(const float* __restrict__ x,
                        const float* __restrict__ agg,
                        const float* __restrict__ deg,
                        const __bf16* __restrict__ Wcat,
                        const float* __restrict__ b_l,
                        const int* __restrict__ batch,
                        float* __restrict__ gsum) {
    __shared__ __bf16 Abuf[16 * AROW];
    __shared__ int sbatch[16];

    const int tid = threadIdx.x;
    const int node0 = blockIdx.x * 16;

    if (tid < 16) sbatch[tid] = batch[node0 + tid];

    // Stage A = [mean_nb | x] for 16 nodes, converted to bf16.
    for (int idx = tid; idx < 16 * KCAT; idx += 128) {
        const int row = idx >> 8;
        const int k   = idx & 255;
        const int node = node0 + row;
        float v;
        if (k < IN_CH) {
            const float d = deg[node];
            v = agg[(size_t)node * IN_CH + k] / fmaxf(d, 1.0f);
        } else {
            v = x[(size_t)node * IN_CH + (k - IN_CH)];
        }
        Abuf[row * AROW + k] = (__bf16)v;
    }
    __syncthreads();

    const int wave = tid >> 5;
    const int lane = tid & 31;
    const int half = lane >> 4;      // selects K sub-block per ISA layout
    const int nIdx = lane & 15;      // B/D column within tile; also A row
    const int o = wave * 16 + nIdx;  // output channel

    v8f c = {};
    #pragma unroll
    for (int ks = 0; ks < 8; ++ks) { // K = 256 in steps of 32
        // A (16-bit 16x32): lane row = nIdx; elements 0..7 -> K=ks*32+half*8+e,
        // elements 8..15 -> K=ks*32+16+half*8+(e-8). Two 16B LDS reads.
        const __bf16* abase = &Abuf[nIdx * AROW + ks * 32 + half * 8];
        const v8bf a0 = *(const v8bf*)abase;
        const v8bf a1 = *(const v8bf*)(abase + 16);
        // B (16-bit 32x16): lane column = nIdx, K same order as A (half*8+e, 16+half*8+(e-8)).
        const __bf16* bbase = &Wcat[(size_t)o * KCAT + ks * 32 + half * 8];
        const v8bf b0 = *(const v8bf*)bbase;
        const v8bf b1 = *(const v8bf*)(bbase + 16);

        v16bf A, B;
        #pragma unroll
        for (int e = 0; e < 8; ++e) {
            A[e] = a0[e]; A[e + 8] = a1[e];
            B[e] = b0[e]; B[e + 8] = b1[e];
        }
        c = __builtin_amdgcn_wmma_f32_16x16x32_bf16(
                false, A, false, B, (short)0, c, false, false);
    }

    // Epilogue: D layout (f32 16x16): VGPR r -> M = r + half*8, N = nIdx.
    const float bl = b_l[o];
    #pragma unroll
    for (int r = 0; r < 8; ++r) {
        const int M = r + half * 8;
        const float h = fmaxf(c[r] + bl, 0.0f);
        atomicAdd(&gsum[sbatch[M] * OUT_CH + o], h);
    }
}

// ---------------------------------------------------------------------------
// Kernel 3: head MLP, one block, fp32 (trivial FLOPs).
// ---------------------------------------------------------------------------
__global__ void k3_head(const float* __restrict__ gsum,
                        const float* __restrict__ gcnt,
                        const float* __restrict__ place,
                        const float* __restrict__ cts,
                        const float* __restrict__ W_p, const float* __restrict__ b_p,
                        const float* __restrict__ W_c, const float* __restrict__ b_c,
                        const float* __restrict__ W_h1, const float* __restrict__ b_h1,
                        const float* __restrict__ W_h2, const float* __restrict__ b_h2,
                        float* __restrict__ out) {
    __shared__ float z[N_GRAPHS * 128];
    __shared__ float h1[N_GRAPHS * 64];
    const int tid = threadIdx.x;

    for (int idx = tid; idx < N_GRAPHS * 128; idx += 256) {
        const int g = idx >> 7, k = idx & 127;
        float v;
        if (k < 64) {
            v = gsum[g * 64 + k] / fmaxf(gcnt[g], 1.0f);
        } else if (k < 96) {
            const int j = k - 64;
            float a = b_p[j];
            for (int t = 0; t < 7; ++t) a += place[g * 7 + t] * W_p[j * 7 + t];
            v = fmaxf(a, 0.0f);
        } else {
            const int j = k - 96;
            float a = b_c[j];
            for (int t = 0; t < 4; ++t) a += cts[g * 4 + t] * W_c[j * 4 + t];
            v = fmaxf(a, 0.0f);
        }
        z[idx] = v;
    }
    __syncthreads();

    for (int idx = tid; idx < N_GRAPHS * 64; idx += 256) {
        const int g = idx >> 6, o = idx & 63;
        float a = b_h1[o];
        for (int k = 0; k < 128; ++k) a += z[g * 128 + k] * W_h1[o * 128 + k];
        h1[idx] = fmaxf(a, 0.0f);
    }
    __syncthreads();

    if (tid < N_GRAPHS) {
        float a = b_h2[0];
        for (int o = 0; o < 64; ++o) a += h1[tid * 64 + o] * W_h2[o];
        out[tid] = a;
    }
}

// ---------------------------------------------------------------------------
// Launch
// ---------------------------------------------------------------------------
extern "C" void kernel_launch(void* const* d_in, const int* in_sizes, int n_in,
                              void* d_out, int out_size, void* d_ws, size_t ws_size,
                              hipStream_t stream) {
    const float* x        = (const float*)d_in[0];
    const float* place    = (const float*)d_in[1];
    const float* cts      = (const float*)d_in[2];
    const float* W_l      = (const float*)d_in[3];
    const float* b_l      = (const float*)d_in[4];
    const float* W_r      = (const float*)d_in[5];
    const float* W_p      = (const float*)d_in[6];
    const float* b_p      = (const float*)d_in[7];
    const float* W_c      = (const float*)d_in[8];
    const float* b_c      = (const float*)d_in[9];
    const float* W_h1     = (const float*)d_in[10];
    const float* b_h1     = (const float*)d_in[11];
    const float* W_h2     = (const float*)d_in[12];
    const float* b_h2     = (const float*)d_in[13];
    const int*   edge_idx = (const int*)d_in[14];
    const int*   batch    = (const int*)d_in[15];

    char* ws = (char*)d_ws;
    float*  agg  = (float*) (ws + OFF_AGG);
    float*  deg  = (float*) (ws + OFF_DEG);
    __bf16* xbf  = (__bf16*)(ws + OFF_XBF);
    __bf16* Wcat = (__bf16*)(ws + OFF_WCAT);
    float*  gsum = (float*) (ws + OFF_GSUM);
    float*  gcnt = (float*) (ws + OFF_GCNT);
    float*  out  = (float*)d_out;

    // 0) init / convert
    {
        const int total = N_NODES * IN_CH;
        k0_init<<<(total + 255) / 256, 256, 0, stream>>>(
            x, W_l, W_r, agg, deg, xbf, Wcat, gsum, gcnt);
    }
    // 1) edge scatter (wave-per-edge, 4 ch per lane)
    {
        const long long total = (long long)N_EDGES * 32;
        k1_edges<<<(unsigned)((total + 255) / 256), 256, 0, stream>>>(
            xbf, edge_idx, agg, deg);
    }
    // 1b) per-graph node counts
    k1b_gcnt<<<(N_NODES + 255) / 256, 256, 0, stream>>>(batch, gcnt);

    // 2) fused WMMA GEMM + ReLU + pool scatter (16 nodes per block, 4 waves)
    k2_wmma<<<N_NODES / 16, 128, 0, stream>>>(
        x, agg, deg, Wcat, b_l, batch, gsum);

    // 3) head MLP
    k3_head<<<1, 256, 0, stream>>>(
        gsum, gcnt, place, cts, W_p, b_p, W_c, b_c, W_h1, b_h1, W_h2, b_h2, out);
}
